// ModifiedCLIPVisionModel_87548613361717
// MI455X (gfx1250) — compile-verified, matrix operand order checked by reading, not run on hardware
//
#include <hip/hip_runtime.h>
#include <hip/hip_bf16.h>
#include <cstdint>
#include <cstddef>

// ---------------------------------------------------------------------------
// ModifiedCLIPVisionModel forward for gfx1250 (MI455X), wave32 + WMMA f16.
// All heavy math runs through v_wmma_f32_16x16x32_f16 with LDS-tiled GEMMs.
// LDS tiles are stored K-contiguous per row/column so every WMMA fragment is
// exactly two ds_load_b128 ops (no scalar u16 packing). Global tile fetch for
// iteration k+1 is issued in registers while iteration k's WMMAs execute.
// MoD top-k block in the reference is an identity copy -> skipped.
// ---------------------------------------------------------------------------

typedef __attribute__((ext_vector_type(16))) _Float16 v16h;
typedef __attribute__((ext_vector_type(8)))  float    v8f;

#define BM 128
#define BN 128
#define BKK 32
#define LDT 40              // LDS row stride in halfs: 80B (16B aligned, bank-spread)

// model dims
#define Dm     1024
#define FFdim  4096
#define Ltot   6
#define Btot   16
#define Stok   257
#define Htot   16
#define SP     288          // S padded to multiple of 32 (for AV GEMM K-dim)
#define Mtok   (Btot*Stok)  // 4112
#define BH     (Btot*Htot)  // 256
#define KP     608          // patch K (3*14*14=588) padded to mult of 32
#define NPatch (Btot*256)   // 4096 patch rows

union H8 { uint4 u; _Float16 h[8]; };

// ---- WMMA fragment load from LDS (two contiguous 8-half runs along K) -----
// Works for both A tiles (row=M) and K-contiguous B tiles (row=N), per the
// CDNA5 ISA 7.12.2 16-bit operand layouts.
__device__ inline v16h frag_ld(const _Float16* base) {
  const int lane = threadIdx.x & 31;
  const _Float16* ptr = base + (lane & 15) * LDT + ((lane >> 4) << 3);
  union { uint4 u[2]; v16h v; } t;
  t.u[0] = *(const uint4*)(ptr);        // k = kh .. kh+7
  t.u[1] = *(const uint4*)(ptr + 16);   // k = 16+kh .. 16+kh+7
  return t.v;
}

// ---- generic batched GEMM:  out = act(scale*A*B + bias) (+ residual) ------
struct GemmP {
  const _Float16* A;  int lda; long long sA1, sA2;
  const _Float16* B;  int ldb; long long sB1, sB2; int transB;
  float*      out32;            // may be null
  _Float16*   out16;            // may be null
  int ldo; long long sO1, sO2;
  const float* bias;            // per-column, may be null
  const float* residual;        // same layout as out, may be null
  int M, N, K, Kvalid;          // K mult of 32; B rows >= Kvalid read as zero
  float scale; int act;         // act=1 -> quick_gelu
  int zdiv;                     // blockIdx.z -> (z/zdiv, z%zdiv) offsets
};

__global__ __launch_bounds__(256)
void gemm_wmma_kernel(GemmP p) {
  __shared__ _Float16 sA [BM * LDT];    // [m][k], k-contiguous
  __shared__ _Float16 sBT[BN * LDT];    // [n][k], k-contiguous

  const int tid = threadIdx.x;
  const int bz  = blockIdx.z;
  const int z1  = bz / p.zdiv, z2 = bz % p.zdiv;
  const _Float16* Ab = p.A + z1 * p.sA1 + z2 * p.sA2;
  const _Float16* Bb = p.B + z1 * p.sB1 + z2 * p.sB2;
  const long long oOff = z1 * p.sO1 + z2 * p.sO2;

  const int m0 = blockIdx.x * BM;
  const int n0 = blockIdx.y * BN;
  const int wave = tid >> 5;
  const int wm = wave >> 2;     // 0..1  (64 rows each)
  const int wn = wave & 3;      // 0..3  (32 cols each)

  v8f acc[4][2];
  v8f vzero = {0.f,0.f,0.f,0.f,0.f,0.f,0.f,0.f};
#pragma unroll
  for (int i = 0; i < 4; ++i)
#pragma unroll
    for (int j = 0; j < 2; ++j) acc[i][j] = vzero;

  // staging registers for the current global tile
  H8 aR[2], bR[2];

  // --- A fetch: thread covers rows idx>>5, 8 k's at idx&31, two passes ---
  auto fetchA = [&](int k0) {
#pragma unroll
    for (int pa = 0; pa < 2; ++pa) {
      int idx = (pa * 256 + tid) * 8;
      int r = idx >> 5, c = idx & 31;
      int gm = m0 + r;
      if (gm < p.M) {
        aR[pa].u = *(const uint4*)(Ab + (long long)gm * p.lda + k0 + c);
      } else {
#pragma unroll
        for (int e = 0; e < 8; ++e) aR[pa].h[e] = (_Float16)0.f;
      }
    }
  };
  auto storeA = [&]() {
#pragma unroll
    for (int pa = 0; pa < 2; ++pa) {
      int idx = (pa * 256 + tid) * 8;
      int r = idx >> 5, c = idx & 31;
      *(uint4*)(sA + r * LDT + c) = aR[pa].u;
    }
  };

  // --- B fetch ---
  // normal: thread -> (n8 = tid&15 -> 8 cols, kp = tid>>4 -> k rows 2kp,2kp+1)
  // transB: thread -> (n = tid&127, ks = (tid>>7)*16 -> 16 k's)
  auto fetchB = [&](int k0) {
    if (!p.transB) {
      int n8 = tid & 15, kp = tid >> 4;
      int gn = n0 + n8 * 8;
#pragma unroll
      for (int q = 0; q < 2; ++q) {
        int gk = k0 + 2 * kp + q;
        if (gk < p.Kvalid && gn + 7 < p.N) {
          bR[q].u = *(const uint4*)(Bb + (long long)gk * p.ldb + gn);
        } else {
#pragma unroll
          for (int e = 0; e < 8; ++e)
            bR[q].h[e] = (gk < p.Kvalid && gn + e < p.N)
                             ? Bb[(long long)gk * p.ldb + gn + e] : (_Float16)0.f;
        }
      }
    } else {
      int n = tid & 127, ks = (tid >> 7) << 4;
      int gn = n0 + n;
#pragma unroll
      for (int q = 0; q < 2; ++q) {
        int kk = k0 + ks + q * 8;
        if (gn < p.N && kk + 7 < p.Kvalid) {
          bR[q].u = *(const uint4*)(Bb + (long long)gn * p.ldb + kk);
        } else {
#pragma unroll
          for (int e = 0; e < 8; ++e)
            bR[q].h[e] = (gn < p.N && kk + e < p.Kvalid)
                             ? Bb[(long long)gn * p.ldb + kk + e] : (_Float16)0.f;
        }
      }
    }
  };
  auto storeB = [&]() {
    if (!p.transB) {
      int n8 = tid & 15, kp = tid >> 4;
#pragma unroll
      for (int e = 0; e < 8; ++e) {
        union { unsigned u; _Float16 h[2]; } t;
        t.h[0] = bR[0].h[e];              // k = 2kp
        t.h[1] = bR[1].h[e];              // k = 2kp+1
        *(unsigned*)(sBT + (n8 * 8 + e) * LDT + 2 * kp) = t.u;
      }
    } else {
      int n = tid & 127, ks = (tid >> 7) << 4;
      *(uint4*)(sBT + n * LDT + ks)     = bR[0].u;
      *(uint4*)(sBT + n * LDT + ks + 8) = bR[1].u;
    }
  };

  fetchA(0);
  fetchB(0);

  for (int k0 = 0; k0 < p.K; k0 += BKK) {
    storeA();
    storeB();
    __syncthreads();
    if (k0 + BKK < p.K) {       // overlap next tile's global loads with WMMA
      fetchA(k0 + BKK);
      fetchB(k0 + BKK);
    }

    v16h af[4], bf[2];
#pragma unroll
    for (int i = 0; i < 4; ++i)
      af[i] = frag_ld(sA + (wm * 64 + i * 16) * LDT);
#pragma unroll
    for (int j = 0; j < 2; ++j)
      bf[j] = frag_ld(sBT + (wn * 32 + j * 16) * LDT);
#pragma unroll
    for (int i = 0; i < 4; ++i)
#pragma unroll
      for (int j = 0; j < 2; ++j)
        acc[i][j] = __builtin_amdgcn_wmma_f32_16x16x32_f16(
            false, af[i], false, bf[j], (short)0, acc[i][j], false, false);
    __syncthreads();
  }

  // --- epilogue ---
  const int lane  = tid & 31;
  const int n_in  = lane & 15;
  const int m_off = (lane >> 4) << 3;
#pragma unroll
  for (int i = 0; i < 4; ++i) {
#pragma unroll
    for (int j = 0; j < 2; ++j) {
#pragma unroll
      for (int r = 0; r < 8; ++r) {
        int gm = m0 + wm * 64 + i * 16 + m_off + r;
        int gn = n0 + wn * 32 + j * 16 + n_in;
        if (gm < p.M && gn < p.N) {
          float v = acc[i][j][r] * p.scale;
          if (p.bias) v += p.bias[gn];
          if (p.act == 1) v = v / (1.f + __expf(-1.702f * v));  // quick_gelu
          long long o = oOff + (long long)gm * p.ldo + gn;
          if (p.residual) v += p.residual[o];
          if (p.out32) p.out32[o] = v;
          if (p.out16) p.out16[o] = (_Float16)v;
        }
      }
    }
  }
}

// ---- LayerNorm: f32 [M,1024] -> f16 normalized ----------------------------
__global__ __launch_bounds__(256)
void layernorm_kernel(const float* __restrict__ x, const float* __restrict__ w,
                      const float* __restrict__ b, _Float16* __restrict__ out) {
  const long long row = blockIdx.x;
  const float* xr = x + row * Dm;
  float lv[4], s = 0.f, s2 = 0.f;
#pragma unroll
  for (int i = 0; i < 4; ++i) {
    float t = xr[threadIdx.x + i * 256];
    lv[i] = t; s += t; s2 += t * t;
  }
#pragma unroll
  for (int off = 16; off; off >>= 1) {
    s  += __shfl_xor(s,  off, 32);
    s2 += __shfl_xor(s2, off, 32);
  }
  __shared__ float ps[8], ps2[8];
  int wave = threadIdx.x >> 5, lane = threadIdx.x & 31;
  if (lane == 0) { ps[wave] = s; ps2[wave] = s2; }
  __syncthreads();
  s = 0.f; s2 = 0.f;
#pragma unroll
  for (int i = 0; i < 8; ++i) { s += ps[i]; s2 += ps2[i]; }
  float mean = s * (1.f / Dm);
  float var  = s2 * (1.f / Dm) - mean * mean;
  float rstd = rsqrtf(var + 1e-5f);
#pragma unroll
  for (int i = 0; i < 4; ++i) {
    int c = threadIdx.x + i * 256;
    out[row * Dm + c] = (_Float16)((lv[i] - mean) * rstd * w[c] + b[c]);
  }
}

// ---- softmax over rows of scores [BH*S, SP], valid cols < Stok ------------
__global__ __launch_bounds__(256)
void softmax_kernel(const float* __restrict__ scores, _Float16* __restrict__ attn,
                    int rows) {
  int wid  = (int)((blockIdx.x * (long long)blockDim.x + threadIdx.x) >> 5);
  int lane = threadIdx.x & 31;
  if (wid >= rows) return;
  const float* src = scores + (long long)wid * SP;
  _Float16*    dst = attn   + (long long)wid * SP;
  float mx = -1e30f;
  for (int i = lane; i < Stok; i += 32) mx = fmaxf(mx, src[i]);
#pragma unroll
  for (int off = 16; off; off >>= 1) mx = fmaxf(mx, __shfl_xor(mx, off, 32));
  float vals[9]; int cnt = 0; float sum = 0.f;
  for (int i = lane; i < Stok; i += 32) {
    float e = __expf(src[i] - mx);
    vals[cnt++] = e; sum += e;
  }
#pragma unroll
  for (int off = 16; off; off >>= 1) sum += __shfl_xor(sum, off, 32);
  float inv = 1.f / sum;
  cnt = 0;
  for (int i = lane; i < Stok; i += 32) dst[i] = (_Float16)(vals[cnt++] * inv);
  for (int i = Stok + lane; i < SP; i += 32) dst[i] = (_Float16)0.f;  // pad
}

// ---- im2col: pixels [B,3,224,224] -> patches f16 [4096, 608] (zero pad) ---
__global__ void im2col_kernel(const float* __restrict__ px,
                              _Float16* __restrict__ patches) {
  long long i = blockIdx.x * (long long)blockDim.x + threadIdx.x;
  if (i >= (long long)NPatch * KP) return;
  int col = (int)(i % KP);
  int row = (int)(i / KP);
  _Float16 v = (_Float16)0.f;
  if (col < 588) {
    int b  = row >> 8, pp = row & 255;
    int py = pp >> 4, pxi = pp & 15;
    int ch = col / 196, rem = col % 196;
    int iy = rem / 14, ix = rem % 14;
    v = (_Float16)px[(((long long)b * 3 + ch) * 224 + (py * 14 + iy)) * 224
                     + (pxi * 14 + ix)];
  }
  patches[i] = v;
}

// ---- assemble x = concat(cls, conv_out) + pos_emb -------------------------
__global__ __launch_bounds__(256)
void assemble_kernel(const float* __restrict__ conv_out,
                     const float* __restrict__ cls,
                     const float* __restrict__ pos, float* __restrict__ x) {
  int row = blockIdx.x;                 // 0..Mtok-1
  int b = row / Stok, s = row % Stok;
#pragma unroll
  for (int i = 0; i < 4; ++i) {
    int c = threadIdx.x + i * 256;
    float v = (s == 0) ? cls[c]
                       : conv_out[((long long)b * 256 + (s - 1)) * Dm + c];
    x[(long long)row * Dm + c] = v + pos[(long long)s * Dm + c];
  }
}

// ---- weight transpose+convert: src f32 [N,K] -> dst f16 [K,ldDst] ---------
__global__ void wconv_kernel(const float* __restrict__ src,
                             _Float16* __restrict__ dst,
                             int N, int K, int ldDst, int colOff) {
  long long i = blockIdx.x * (long long)blockDim.x + threadIdx.x;
  if (i >= (long long)N * K) return;
  int n = (int)(i / K), k = (int)(i % K);
  dst[(long long)k * ldDst + colOff + n] = (_Float16)src[i];
}

__global__ void fill0h_kernel(_Float16* p, long long n) {
  long long i = blockIdx.x * (long long)blockDim.x + threadIdx.x;
  if (i < n) p[i] = (_Float16)0.f;
}

__global__ void biascat_kernel(const float* qb, const float* kb,
                               const float* vb, float* dst) {
  long long i = blockIdx.x * (long long)blockDim.x + threadIdx.x;
  if (i >= (long long)Ltot * 3 * Dm) return;
  int l = (int)(i / (3 * Dm)), c = (int)(i % (3 * Dm));
  float v = (c < Dm) ? qb[l * Dm + c]
                     : (c < 2 * Dm ? kb[l * Dm + c - Dm] : vb[l * Dm + c - 2 * Dm]);
  dst[i] = v;
}

// ---------------------------------------------------------------------------
static inline size_t alignup(size_t x) { return (x + 255) & ~(size_t)255; }

extern "C" void kernel_launch(void* const* d_in, const int* in_sizes, int n_in,
                              void* d_out, int out_size, void* d_ws, size_t ws_size,
                              hipStream_t stream) {
  (void)in_sizes; (void)n_in; (void)out_size; (void)ws_size;
  const float* pixel   = (const float*)d_in[0];
  const float* patch_w = (const float*)d_in[1];
  const float* cls     = (const float*)d_in[2];
  const float* pos     = (const float*)d_in[3];
  const float* ln1_w   = (const float*)d_in[4];
  const float* ln1_b   = (const float*)d_in[5];
  const float* q_w     = (const float*)d_in[6];
  const float* q_b     = (const float*)d_in[7];
  const float* k_w     = (const float*)d_in[8];
  const float* k_b     = (const float*)d_in[9];
  const float* v_w     = (const float*)d_in[10];
  const float* v_b     = (const float*)d_in[11];
  const float* o_w     = (const float*)d_in[12];
  const float* o_b     = (const float*)d_in[13];
  const float* ln2_w   = (const float*)d_in[14];
  const float* ln2_b   = (const float*)d_in[15];
  const float* fc1_w   = (const float*)d_in[16];
  const float* fc1_b   = (const float*)d_in[17];
  const float* fc2_w   = (const float*)d_in[18];
  const float* fc2_b   = (const float*)d_in[19];

  // ---- workspace carve-up ----
  char* w = (char*)d_ws;
  auto take = [&](size_t bytes) { void* p = w; w += alignup(bytes); return p; };
  _Float16* wqkv  = (_Float16*)take((size_t)Ltot * Dm * 3 * Dm * 2);
  _Float16* wo    = (_Float16*)take((size_t)Ltot * Dm * Dm * 2);
  _Float16* wfc1  = (_Float16*)take((size_t)Ltot * Dm * FFdim * 2);
  _Float16* wfc2  = (_Float16*)take((size_t)Ltot * FFdim * Dm * 2);
  float*    qkvb  = (float*)   take((size_t)Ltot * 3 * Dm * 4);
  _Float16* pwT   = (_Float16*)take((size_t)KP * Dm * 2);
  _Float16* patch = (_Float16*)take((size_t)NPatch * KP * 2);
  float*    convo = (float*)   take((size_t)NPatch * Dm * 4);
  float*    xbuf  = (float*)   take((size_t)Mtok * Dm * 4);
  _Float16* h16   = (_Float16*)take((size_t)Mtok * Dm * 2);
  _Float16* qkv   = (_Float16*)take((size_t)Mtok * 3 * Dm * 2);
  float*    score = (float*)   take((size_t)BH * Stok * SP * 4);
  _Float16* attn  = (_Float16*)take((size_t)BH * Stok * SP * 2);
  _Float16* ctx16 = (_Float16*)take((size_t)Mtok * Dm * 2);
  _Float16* ff16  = (_Float16*)take((size_t)Mtok * FFdim * 2);

  auto blocks = [](long long n) { return (unsigned)((n + 255) / 256); };

  // ---- per-launch weight prep (deterministic every call) ----
  fill0h_kernel<<<blocks((long long)KP * Dm), 256, 0, stream>>>(pwT, (long long)KP * Dm);
  wconv_kernel<<<blocks((long long)Dm * 588), 256, 0, stream>>>(patch_w, pwT, Dm, 588, Dm, 0);
  for (int l = 0; l < Ltot; ++l) {
    wconv_kernel<<<blocks((long long)Dm * Dm), 256, 0, stream>>>(
        q_w + (size_t)l * Dm * Dm, wqkv + (size_t)l * Dm * 3 * Dm, Dm, Dm, 3 * Dm, 0);
    wconv_kernel<<<blocks((long long)Dm * Dm), 256, 0, stream>>>(
        k_w + (size_t)l * Dm * Dm, wqkv + (size_t)l * Dm * 3 * Dm, Dm, Dm, 3 * Dm, Dm);
    wconv_kernel<<<blocks((long long)Dm * Dm), 256, 0, stream>>>(
        v_w + (size_t)l * Dm * Dm, wqkv + (size_t)l * Dm * 3 * Dm, Dm, Dm, 3 * Dm, 2 * Dm);
    wconv_kernel<<<blocks((long long)Dm * Dm), 256, 0, stream>>>(
        o_w + (size_t)l * Dm * Dm, wo + (size_t)l * Dm * Dm, Dm, Dm, Dm, 0);
    wconv_kernel<<<blocks((long long)FFdim * Dm), 256, 0, stream>>>(
        fc1_w + (size_t)l * FFdim * Dm, wfc1 + (size_t)l * Dm * FFdim, FFdim, Dm, FFdim, 0);
    wconv_kernel<<<blocks((long long)Dm * FFdim), 256, 0, stream>>>(
        fc2_w + (size_t)l * Dm * FFdim, wfc2 + (size_t)l * FFdim * Dm, Dm, FFdim, Dm, 0);
  }
  biascat_kernel<<<blocks((long long)Ltot * 3 * Dm), 256, 0, stream>>>(q_b, k_b, v_b, qkvb);

  // ---- patch embedding: im2col + GEMM + assemble ----
  im2col_kernel<<<blocks((long long)NPatch * KP), 256, 0, stream>>>(pixel, patch);
  {
    GemmP p{};
    p.A = patch; p.lda = KP;    p.sA1 = 0; p.sA2 = 0;
    p.B = pwT;   p.ldb = Dm;    p.sB1 = 0; p.sB2 = 0; p.transB = 0;
    p.out32 = convo; p.out16 = nullptr; p.ldo = Dm; p.sO1 = 0; p.sO2 = 0;
    p.bias = nullptr; p.residual = nullptr;
    p.M = NPatch; p.N = Dm; p.K = KP; p.Kvalid = KP; p.scale = 1.f; p.act = 0; p.zdiv = 1;
    gemm_wmma_kernel<<<dim3((NPatch + BM - 1) / BM, (Dm + BN - 1) / BN, 1), 256, 0, stream>>>(p);
  }
  assemble_kernel<<<Mtok, 256, 0, stream>>>(convo, cls, pos, xbuf);

  const long long SS = (long long)Stok * SP;   // per-(b,h) score stride
  const float attn_scale = 0.125f;             // DH^-0.5, DH=64

  for (int l = 0; l < Ltot; ++l) {
    // LN1
    layernorm_kernel<<<Mtok, 256, 0, stream>>>(xbuf, ln1_w + l * Dm, ln1_b + l * Dm, h16);
    // QKV projection (fused N = 3072)
    {
      GemmP p{};
      p.A = h16; p.lda = Dm; p.B = wqkv + (size_t)l * Dm * 3 * Dm; p.ldb = 3 * Dm;
      p.transB = 0; p.out16 = qkv; p.out32 = nullptr; p.ldo = 3 * Dm;
      p.bias = qkvb + (size_t)l * 3 * Dm; p.residual = nullptr;
      p.M = Mtok; p.N = 3 * Dm; p.K = Dm; p.Kvalid = Dm; p.scale = 1.f; p.act = 0; p.zdiv = 1;
      gemm_wmma_kernel<<<dim3((Mtok + BM - 1) / BM, (3 * Dm + BN - 1) / BN, 1), 256, 0, stream>>>(p);
    }
    // scores = scale * Q K^T   (batched over B*H, transB)
    {
      GemmP p{};
      p.A = qkv;           p.lda = 3 * Dm; p.sA1 = (long long)Stok * 3 * Dm; p.sA2 = 64;
      p.B = qkv + Dm;      p.ldb = 3 * Dm; p.sB1 = (long long)Stok * 3 * Dm; p.sB2 = 64;
      p.transB = 1;
      p.out32 = score; p.out16 = nullptr; p.ldo = SP;
      p.sO1 = (long long)Htot * SS; p.sO2 = SS;
      p.bias = nullptr; p.residual = nullptr;
      p.M = Stok; p.N = Stok; p.K = 64; p.Kvalid = 64;
      p.scale = attn_scale; p.act = 0; p.zdiv = Htot;
      gemm_wmma_kernel<<<dim3(3, 3, BH), 256, 0, stream>>>(p);
    }
    // softmax rows, f16 output with zero padding to SP
    softmax_kernel<<<(BH * Stok + 7) / 8, 256, 0, stream>>>(score, attn, BH * Stok);
    // ctx = attn @ V  (K padded to 288, Kvalid=257)
    {
      GemmP p{};
      p.A = attn;          p.lda = SP; p.sA1 = (long long)Htot * SS; p.sA2 = SS;
      p.B = qkv + 2 * Dm;  p.ldb = 3 * Dm; p.sB1 = (long long)Stok * 3 * Dm; p.sB2 = 64;
      p.transB = 0;
      p.out16 = ctx16; p.out32 = nullptr; p.ldo = Dm;
      p.sO1 = (long long)Stok * Dm; p.sO2 = 64;
      p.bias = nullptr; p.residual = nullptr;
      p.M = Stok; p.N = 64; p.K = SP; p.Kvalid = Stok;
      p.scale = 1.f; p.act = 0; p.zdiv = Htot;
      gemm_wmma_kernel<<<dim3(3, 1, BH), 256, 0, stream>>>(p);
    }
    // O projection + residual -> x (f32)
    {
      GemmP p{};
      p.A = ctx16; p.lda = Dm; p.B = wo + (size_t)l * Dm * Dm; p.ldb = Dm; p.transB = 0;
      p.out32 = xbuf; p.out16 = nullptr; p.ldo = Dm;
      p.bias = o_b + (size_t)l * Dm; p.residual = xbuf;
      p.M = Mtok; p.N = Dm; p.K = Dm; p.Kvalid = Dm; p.scale = 1.f; p.act = 0; p.zdiv = 1;
      gemm_wmma_kernel<<<dim3((Mtok + BM - 1) / BM, (Dm + BN - 1) / BN, 1), 256, 0, stream>>>(p);
    }
    // LN2
    layernorm_kernel<<<Mtok, 256, 0, stream>>>(xbuf, ln2_w + l * Dm, ln2_b + l * Dm, h16);
    // FC1 + quick_gelu -> ff16
    {
      GemmP p{};
      p.A = h16; p.lda = Dm; p.B = wfc1 + (size_t)l * Dm * FFdim; p.ldb = FFdim; p.transB = 0;
      p.out16 = ff16; p.out32 = nullptr; p.ldo = FFdim;
      p.bias = fc1_b + (size_t)l * FFdim; p.residual = nullptr;
      p.M = Mtok; p.N = FFdim; p.K = Dm; p.Kvalid = Dm; p.scale = 1.f; p.act = 1; p.zdiv = 1;
      gemm_wmma_kernel<<<dim3((Mtok + BM - 1) / BM, (FFdim + BN - 1) / BN, 1), 256, 0, stream>>>(p);
    }
    // FC2 + residual -> x (f32)
    {
      GemmP p{};
      p.A = ff16; p.lda = FFdim; p.B = wfc2 + (size_t)l * FFdim * Dm; p.ldb = Dm; p.transB = 0;
      p.out32 = xbuf; p.out16 = nullptr; p.ldo = Dm;
      p.bias = fc2_b + (size_t)l * Dm; p.residual = xbuf;
      p.M = Mtok; p.N = Dm; p.K = FFdim; p.Kvalid = FFdim; p.scale = 1.f; p.act = 0; p.zdiv = 1;
      gemm_wmma_kernel<<<dim3((Mtok + BM - 1) / BM, (Dm + BN - 1) / BN, 1), 256, 0, stream>>>(p);
    }
  }

  hipMemcpyAsync(d_out, xbuf, (size_t)Mtok * Dm * sizeof(float),
                 hipMemcpyDeviceToDevice, stream);
}
